// Attention_59459527246136
// MI455X (gfx1250) — compile-verified, matrix operand order checked by reading, not run on hardware
//
#include <hip/hip_runtime.h>
#include <hip/hip_bf16.h>
#include <cstdint>
#include <cstddef>

// ---------------------------------------------------------------------------
// MI455X (gfx1250) fused attention pipeline, bf16 WMMA GEMMs with async
// LDS staging (GLOBAL_LOAD_ASYNC_TO_LDS_B128 + ASYNCcnt, triple buffered).
//   B=8192, T=3, D=2048.  ~687 GFLOP -> v_wmma_f32_16x16x32_bf16.
//   Output projection folded: sum_t(x_t@wo^T+bo) = (sum_t x_t)@wo^T + 3*bo.
// ---------------------------------------------------------------------------

typedef __bf16 bf16;
typedef __bf16 v16bf __attribute__((ext_vector_type(16)));
typedef __bf16 v8bf  __attribute__((ext_vector_type(8)));
typedef __bf16 v4bf  __attribute__((ext_vector_type(4)));
typedef float  v8f   __attribute__((ext_vector_type(8)));
typedef float  v4f   __attribute__((ext_vector_type(4)));

#define B_   8192
#define T_   3
#define D_   2048
#define M1_  (B_ * T_)          /* 24576 rows of Q/K/V */
#define ZROW (2 * T_ * D_)      /* 12288 floats per z row */

// GEMM tiling
#define TM   128                /* WG tile M   */
#define TN   256                /* WG tile N   */
#define TK   32                 /* K step (one WMMA K) */
#define ABYTES (TM * TK * 2)    /* 8192  B  = 512 x 16B chunks  */
#define BBYTES (TN * TK * 2)    /* 16384 B  = 1024 x 16B chunks */
#define BUFBYTES (ABYTES + BBYTES)  /* 24576 B */
#define NBUF 3

// ---------------------------------------------------------------------------
// Kernel 0a: gather+convert z -> contiguous bf16 zq (24576x2048), zk (same)
// ---------------------------------------------------------------------------
__global__ __launch_bounds__(256) void convert_z_kernel(
    const float* __restrict__ z, bf16* __restrict__ zq, bf16* __restrict__ zk) {
  size_t vi = (size_t)blockIdx.x * blockDim.x + threadIdx.x;  // vec-4 index
  size_t e  = vi * 4;
  if (e >= (size_t)M1_ * D_) return;
  int m = (int)(e >> 11);
  int k = (int)(e & 2047);
  int b = m / 3;
  int t = m - 3 * b;
  const float* srcq = z + (size_t)b * ZROW + (size_t)t * D_ + k;
  const float* srck = srcq + T_ * D_;
  v4f q4 = *(const v4f*)srcq;
  v4f k4 = *(const v4f*)srck;
  v4bf qo, ko;
#pragma unroll
  for (int i = 0; i < 4; ++i) { qo[i] = (bf16)q4[i]; ko[i] = (bf16)k4[i]; }
  *(v4bf*)(zq + e) = qo;
  *(v4bf*)(zk + e) = ko;
}

// ---------------------------------------------------------------------------
// Kernel 0b: elementwise f32 -> bf16 (weights)
// ---------------------------------------------------------------------------
__global__ __launch_bounds__(256) void convert_f32_bf16_kernel(
    const float* __restrict__ src, bf16* __restrict__ dst, size_t n4) {
  size_t vi = (size_t)blockIdx.x * blockDim.x + threadIdx.x;
  if (vi >= n4) return;
  size_t e = vi * 4;
  v4f s = *(const v4f*)(src + e);
  v4bf d;
#pragma unroll
  for (int i = 0; i < 4; ++i) d[i] = (bf16)s[i];
  *(v4bf*)(dst + e) = d;
}

// ---------------------------------------------------------------------------
// WMMA bf16 GEMM with async-LDS staging.
//   C[M,N] = A[M,K] @ W[N,K]^T + bias_scale*bias[N]
// LDS tile layout is chunk-major: 16-byte chunk (c,row) lives at
//   A: 16*(c*TM + row)   c in 0..3 (k-chunk of 8 bf16), row in 0..TM-1
//   B: ABYTES + 16*(c*TN + row)
// -> per-thread async-copy map and per-lane fragment reads (ds_load_b128)
// are trivial linear functions of tid/lane.  Fragment per CDNA5 16-bit A
// layout: lane L holds row (L&15); k-chunks c=(L>>4) and c+2.
// Wave tile 64Mx64N (16 accums); 8 waves -> 128M x 256N per WG.
// Pipeline: triple buffer, one split-barrier per K-step; last two steps
// peeled so the steady loop has a single unconditional s_wait_asynccnt.
// ---------------------------------------------------------------------------
__device__ __forceinline__ v16bf ld_frag_lds(const char* base, int lo, int hi) {
  v8bf c0 = *(const v8bf*)(base + 16 * lo);
  v8bf c1 = *(const v8bf*)(base + 16 * hi);
  return __builtin_shufflevector(c0, c1, 0, 1, 2, 3, 4, 5, 6, 7,
                                 8, 9, 10, 11, 12, 13, 14, 15);
}

template <bool OUT_F32>
__global__ __launch_bounds__(256) void gemm_bf16_kernel(
    const bf16* __restrict__ A, const bf16* __restrict__ W,
    const float* __restrict__ bias, float bias_scale,
    void* __restrict__ Cout, int M, int N, int K) {
  __shared__ char ldsbuf[NBUF * BUFBYTES];   // 72 KB (WGP has 320 KB)

  const int tid   = threadIdx.x;
  const int lane  = tid & 31;
  const int wave  = tid >> 5;
  const int m0w   = (wave & 1) * 64;         // wave offset in WG tile (M)
  const int n0w   = (wave >> 1) * 64;        // wave offset in WG tile (N)
  const int lrow  = lane & 15;
  const int clo   = lane >> 4;               // lane's low k-chunk (0 or 1)
  const int khalf = clo << 3;                // C-layout M offset (0 or 8)
  const int tileM = blockIdx.y * TM;
  const int tileN = blockIdx.x * TN;

  const unsigned ldsBase = (unsigned)(uintptr_t)(void*)ldsbuf;

  // ---- per-thread async-copy assignments (6 x B128 per thread per step) ----
  const bf16* gcur[6];
  unsigned    loff[6];
  {
    int qa0 = tid, qa1 = tid + 256;                    // A: 512 chunks
    gcur[0] = A + (size_t)(tileM + (qa0 & (TM - 1))) * K + (qa0 >> 7) * 8;
    loff[0] = 16u * (unsigned)qa0;
    gcur[1] = A + (size_t)(tileM + (qa1 & (TM - 1))) * K + (qa1 >> 7) * 8;
    loff[1] = 16u * (unsigned)qa1;
#pragma unroll
    for (int j = 0; j < 4; ++j) {                      // B: 1024 chunks
      int q = tid + 256 * j;
      gcur[2 + j] = W + (size_t)(tileN + (q & (TN - 1))) * K + (q >> 8) * 8;
      loff[2 + j] = (unsigned)ABYTES + 16u * (unsigned)q;
    }
  }

  auto issue = [&](int buf) {
    unsigned boff = ldsBase + (unsigned)buf * (unsigned)BUFBYTES;
#pragma unroll
    for (int u = 0; u < 6; ++u) {
      asm volatile("global_load_async_to_lds_b128 %0, %1, off"
                   :: "v"(boff + loff[u]),
                      "v"((unsigned long long)(uintptr_t)gcur[u])
                   : "memory");
      gcur[u] += TK;
    }
  };

  v8f acc[4][4] = {};

  auto compute = [&](int buf) {
    const char* abuf = ldsbuf + buf * BUFBYTES;
    const char* bbuf = abuf + ABYTES;
    v16bf af[4], bfr[4];
#pragma unroll
    for (int i = 0; i < 4; ++i) {
      int r = m0w + i * 16 + lrow;
      af[i] = ld_frag_lds(abuf, clo * TM + r, (clo + 2) * TM + r);
    }
#pragma unroll
    for (int j = 0; j < 4; ++j) {
      int r = n0w + j * 16 + lrow;
      bfr[j] = ld_frag_lds(bbuf, clo * TN + r, (clo + 2) * TN + r);
    }
#pragma unroll
    for (int i = 0; i < 4; ++i)
#pragma unroll
      for (int j = 0; j < 4; ++j)
        acc[i][j] = __builtin_amdgcn_wmma_f32_16x16x32_bf16(
            false, af[i], false, bfr[j], (short)0, acc[i][j], false, false);
  };

  const int nsteps = K >> 5;   // K = 2048 -> 64 steps (assumed >= 2)

  issue(0);
  issue(1);

  int bufC = 0;                // buffer computed this step
  int bufI = 2;                // buffer filled for step s+2
  for (int s = 0; s < nsteps - 2; ++s) {
    // my step-s loads done (ASYNCcnt in-order; <=6 leaves step s+1 in flight)
    asm volatile("s_wait_asynccnt 0x6" ::: "memory");
    __syncthreads();           // everyone's step-s data visible; s-1 buf free
    issue(bufI);
    bufI = (bufI == NBUF - 1) ? 0 : bufI + 1;
    compute(bufC);
    bufC = (bufC == NBUF - 1) ? 0 : bufC + 1;
  }
  // step nsteps-2: nothing left to issue
  asm volatile("s_wait_asynccnt 0x6" ::: "memory");
  __syncthreads();
  compute(bufC);
  bufC = (bufC == NBUF - 1) ? 0 : bufC + 1;
  // step nsteps-1: drain
  asm volatile("s_wait_asynccnt 0x0" ::: "memory");
  __syncthreads();
  compute(bufC);

  // C layout (32-bit C/D 16x16): VGPR r -> M = r + khalf; lane%16 -> N.
#pragma unroll
  for (int i = 0; i < 4; ++i) {
#pragma unroll
    for (int j = 0; j < 4; ++j) {
      const int nn = tileN + n0w + j * 16 + lrow;
      const float bv = bias_scale * bias[nn];
#pragma unroll
      for (int r = 0; r < 8; ++r) {
        const int mm  = tileM + m0w + i * 16 + khalf + r;
        const float v = acc[i][j][r] + bv;
        if (OUT_F32) ((float*)Cout)[(size_t)mm * N + nn] = v;
        else         ((bf16*)Cout)[(size_t)mm * N + nn] = (bf16)v;
      }
    }
  }
}

// ---------------------------------------------------------------------------
// Kernel 2: per-batch attention (T=3). One 256-thread block per batch.
// scores[t][s] = (t>s) ? 0 : (q_t . k_s)/sqrt(D); softmax over s (zeros kept);
// xs[b] = sum_s (sum_t p[t][s]) * v_s  (bf16 out, feeds output GEMM).
// ---------------------------------------------------------------------------
__global__ __launch_bounds__(256) void attn_kernel(
    const bf16* __restrict__ Q, const bf16* __restrict__ Kb,
    const bf16* __restrict__ Vb, bf16* __restrict__ XS) {
  __shared__ float partial[9][8];
  __shared__ float cs[3];

  const int b   = blockIdx.x;
  const int tid = threadIdx.x;
  const size_t base = (size_t)b * 3 * D_;

  float dot[9] = {0, 0, 0, 0, 0, 0, 0, 0, 0};
  for (int k = tid; k < D_; k += 256) {
    float q0 = (float)Q[base + k];
    float q1 = (float)Q[base + D_ + k];
    float q2 = (float)Q[base + 2 * D_ + k];
    float c0 = (float)Kb[base + k];
    float c1 = (float)Kb[base + D_ + k];
    float c2 = (float)Kb[base + 2 * D_ + k];
    dot[0] += q0 * c0; dot[1] += q0 * c1; dot[2] += q0 * c2;
    dot[3] += q1 * c0; dot[4] += q1 * c1; dot[5] += q1 * c2;
    dot[6] += q2 * c0; dot[7] += q2 * c1; dot[8] += q2 * c2;
  }
#pragma unroll
  for (int i = 0; i < 9; ++i)
    for (int off = 16; off > 0; off >>= 1) dot[i] += __shfl_down(dot[i], off);
  const int wv = tid >> 5, ln = tid & 31;
  if (ln == 0)
#pragma unroll
    for (int i = 0; i < 9; ++i) partial[i][wv] = dot[i];
  __syncthreads();

  if (tid == 0) {
    const float scale = 0.022097086912079608f;  // 1/sqrt(2048)
    float s[3][3];
    for (int t = 0; t < 3; ++t)
      for (int ss = 0; ss < 3; ++ss) {
        float d = 0.f;
        for (int w = 0; w < 8; ++w) d += partial[t * 3 + ss][w];
        s[t][ss] = (t > ss) ? 0.0f : d * scale;   // mask BEFORE softmax, ->0.0
      }
    float c[3] = {0.f, 0.f, 0.f};
    for (int t = 0; t < 3; ++t) {
      float m = fmaxf(s[t][0], fmaxf(s[t][1], s[t][2]));
      float e0 = expf(s[t][0] - m), e1 = expf(s[t][1] - m), e2 = expf(s[t][2] - m);
      float inv = 1.0f / (e0 + e1 + e2);
      c[0] += e0 * inv; c[1] += e1 * inv; c[2] += e2 * inv;
    }
    cs[0] = c[0]; cs[1] = c[1]; cs[2] = c[2];
  }
  __syncthreads();

  const float c0 = cs[0], c1 = cs[1], c2 = cs[2];
  for (int k = tid; k < D_; k += 256) {
    float v = c0 * (float)Vb[base + k] + c1 * (float)Vb[base + D_ + k] +
              c2 * (float)Vb[base + 2 * D_ + k];
    XS[(size_t)b * D_ + k] = (bf16)v;
  }
}

// ---------------------------------------------------------------------------
// Launcher
// ---------------------------------------------------------------------------
extern "C" void kernel_launch(void* const* d_in, const int* in_sizes, int n_in,
                              void* d_out, int out_size, void* d_ws, size_t ws_size,
                              hipStream_t stream) {
  const float* z  = (const float*)d_in[0];
  const float* wq = (const float*)d_in[1];
  const float* bq = (const float*)d_in[2];
  const float* wk = (const float*)d_in[3];
  const float* bk = (const float*)d_in[4];
  const float* wv = (const float*)d_in[5];
  const float* bv = (const float*)d_in[6];
  const float* wo = (const float*)d_in[7];
  const float* bo = (const float*)d_in[8];
  float* out = (float*)d_out;

  char* ws = (char*)d_ws;
  size_t off = 0;
  auto carve = [&](size_t bytes) -> char* {
    char* p = ws + off;
    off += (bytes + 255) & ~(size_t)255;
    return p;
  };
  bf16* zq  = (bf16*)carve((size_t)M1_ * D_ * 2);
  bf16* zk  = (bf16*)carve((size_t)M1_ * D_ * 2);
  bf16* wqb = (bf16*)carve((size_t)D_ * D_ * 2);
  bf16* wkb = (bf16*)carve((size_t)D_ * D_ * 2);
  bf16* wvb = (bf16*)carve((size_t)D_ * D_ * 2);
  bf16* wob = (bf16*)carve((size_t)D_ * D_ * 2);
  bf16* Qb  = (bf16*)carve((size_t)M1_ * D_ * 2);
  bf16* Kc  = (bf16*)carve((size_t)M1_ * D_ * 2);
  bf16* Vc  = (bf16*)carve((size_t)M1_ * D_ * 2);
  bf16* xs  = (bf16*)carve((size_t)B_ * D_ * 2);
  (void)ws_size; (void)in_sizes; (void)n_in; (void)out_size;

  // 0) convert/pack to bf16
  {
    size_t n4 = (size_t)M1_ * D_ / 4;
    int grid = (int)((n4 + 255) / 256);
    convert_z_kernel<<<grid, 256, 0, stream>>>(z, zq, zk);
  }
  {
    size_t n4 = (size_t)D_ * D_ / 4;
    int grid = (int)((n4 + 255) / 256);
    convert_f32_bf16_kernel<<<grid, 256, 0, stream>>>(wq, wqb, n4);
    convert_f32_bf16_kernel<<<grid, 256, 0, stream>>>(wk, wkb, n4);
    convert_f32_bf16_kernel<<<grid, 256, 0, stream>>>(wv, wvb, n4);
    convert_f32_bf16_kernel<<<grid, 256, 0, stream>>>(wo, wob, n4);
  }

  // 1) Q/K/V projections (bf16 WMMA, f32 accumulate, bf16 out)
  dim3 blk(256);
  dim3 g1(D_ / TN, M1_ / TM);    // (8, 192)
  gemm_bf16_kernel<false><<<g1, blk, 0, stream>>>(zq, wqb, bq, 1.0f, Qb, M1_, D_, D_);
  gemm_bf16_kernel<false><<<g1, blk, 0, stream>>>(zk, wkb, bk, 1.0f, Kc, M1_, D_, D_);
  gemm_bf16_kernel<false><<<g1, blk, 0, stream>>>(zq, wvb, bv, 1.0f, Vc, M1_, D_, D_);

  // 2) per-batch attention -> pre-summed xs (bf16)
  attn_kernel<<<B_, 256, 0, stream>>>(Qb, Kc, Vc, xs);

  // 3) output projection with folded T-sum: out = xs @ wo^T + 3*bo (f32 out)
  dim3 g2(D_ / TN, B_ / TM);     // (8, 64)
  gemm_bf16_kernel<true><<<g2, blk, 0, stream>>>(xs, wob, bo, 3.0f, out, B_, D_, D_);
}
